// CrossAttention_59322088292884
// MI455X (gfx1250) — compile-verified, matrix operand order checked by reading, not run on hardware
//
#include <hip/hip_runtime.h>
#include <stdint.h>

// Problem constants (from reference)
constexpr int B_ = 2, N_ = 2048, M_ = 2048, D_ = 1024, H_ = 16, HD = 64;
constexpr float EPS = 1e-6f;

typedef __attribute__((ext_vector_type(16))) _Float16 v16h;
typedef __attribute__((ext_vector_type(8)))  float    v8f;

union Frag16 {
    v16h     v;
    uint32_t u[8];
};

// Load a 16x32 f16 fragment (A or B operand of V_WMMA_F32_16X16X32_F16).
// rowptr = base of this lane's row (m or n index already applied), K-contiguous.
// hi = lane>>4. Layout per CDNA5 ISA 7.12.2: dword j holds K pair
// (j>=4?16:0) + hi*8 + (j&3)*2  -> two contiguous 16B halves (b128-mergeable).
__device__ __forceinline__ void load_frag(Frag16& f, const _Float16* rowptr, int hi) {
#pragma unroll
    for (int j = 0; j < 8; ++j) {
        const int ko = ((j >= 4) ? 16 : 0) + hi * 8 + (j & 3) * 2;
        f.u[j] = *reinterpret_cast<const uint32_t*>(rowptr + ko);
    }
}

__device__ __forceinline__ v8f wmma_f16(const Frag16& a, const Frag16& b, v8f c) {
    return __builtin_amdgcn_wmma_f32_16x16x32_f16(false, a.v, false, b.v,
                                                  (short)0, c, false, false);
}

// Intra-wave LDS store->load ordering: DS pipeline is in-order per wave on
// CDNA5; we only need a compiler-level fence so the ds_load is not hoisted
// above the cross-lane ds_stores. No hardware barrier needed (region is
// wave-private).
__device__ __forceinline__ void wave_lds_fence() {
    __builtin_amdgcn_fence(__ATOMIC_ACQ_REL, "wavefront");
    __builtin_amdgcn_wave_barrier();
}

// ---------------------------------------------------------------------------
// Elementwise f32 -> f16
// ---------------------------------------------------------------------------
__global__ void cvt_f32_f16(const float* __restrict__ in, _Float16* __restrict__ out,
                            size_t n) {
    size_t i = (size_t)blockIdx.x * blockDim.x + threadIdx.x;
    const size_t stride = (size_t)gridDim.x * blockDim.x;
    for (; i < n; i += stride) out[i] = (_Float16)in[i];
}

// ---------------------------------------------------------------------------
// V projection: f32 [b][m][d] -> f16 transposed [ (b*H+h)*HD + c ][ m ]
// so that P@V B-operand has contiguous K (=key) pairs.
// ---------------------------------------------------------------------------
__global__ void v_transpose_cvt(const float* __restrict__ in, _Float16* __restrict__ out) {
    const size_t total = (size_t)B_ * M_ * D_;
    size_t i = (size_t)blockIdx.x * blockDim.x + threadIdx.x;
    const size_t stride = (size_t)gridDim.x * blockDim.x;
    for (; i < total; i += stride) {
        const int d = (int)(i % D_);
        const size_t bm = i / D_;
        const int m = (int)(bm % M_);
        const int b = (int)(bm / M_);
        const int h = d >> 6, c = d & 63;
        out[(((size_t)b * H_ + h) * HD + c) * M_ + m] = (_Float16)in[i];
    }
}

// ---------------------------------------------------------------------------
// Row-wise RMSNorm(+g,+scale) f32 -> f16. One block per row of Dd elements.
// ---------------------------------------------------------------------------
__global__ void rmsnorm_cvt(const float* __restrict__ in, const float* __restrict__ g,
                            float scale, _Float16* __restrict__ out, int Dd) {
    const int row = blockIdx.x;
    const float* x = in + (size_t)row * Dd;
    float ss = 0.f;
    for (int i = threadIdx.x; i < Dd; i += blockDim.x) {
        const float v = x[i];
        ss += v * v;
    }
    __shared__ float red[256];
    red[threadIdx.x] = ss;
    __syncthreads();
    for (int s = 128; s > 0; s >>= 1) {
        if ((int)threadIdx.x < s) red[threadIdx.x] += red[threadIdx.x + s];
        __syncthreads();
    }
    const float f = rsqrtf(red[0] / (float)Dd + EPS) * scale;
    for (int i = threadIdx.x; i < Dd; i += blockDim.x)
        out[(size_t)row * Dd + i] = (_Float16)(x[i] * f * g[i]);
}

// ---------------------------------------------------------------------------
// WMMA GEMM: C[R x Cn] = A[R x K] * W[Cn x K]^T (+ bias[Cn]), f16 in / f32 out.
// Block = 8 waves; wave computes 32x64 tile; block tile 128x128.
// R, Cn multiples of 128; K multiple of 32.
// ---------------------------------------------------------------------------
__global__ __launch_bounds__(256)
void gemm_nt_wmma(const _Float16* __restrict__ A, const _Float16* __restrict__ W,
                  const float* __restrict__ bias, float* __restrict__ C,
                  int R, int K, int Cn) {
    const int lane = threadIdx.x & 31;
    const int wave = threadIdx.x >> 5;
    const int m16 = lane & 15, hi = lane >> 4;
    const int rw = wave & 3, cw = wave >> 2;

    const int row0 = blockIdx.x * 128 + rw * 32;  // wave: rows [row0, row0+32)
    const int col0 = blockIdx.y * 128 + cw * 64;  // wave: cols [col0, col0+64)
    (void)R;

    const _Float16* arow0 = A + (size_t)(row0 + m16) * K;
    const _Float16* arow1 = A + (size_t)(row0 + 16 + m16) * K;

    // One-shot warm-up prefetch (global_prefetch_b8); outside the hot loop so
    // it cannot break the WMMA pipeline or trigger hazard NOPs.
    __builtin_prefetch(arow0, 0, 3);
    __builtin_prefetch(arow1, 0, 3);
#pragma unroll
    for (int t = 0; t < 4; ++t)
        __builtin_prefetch(W + (size_t)(col0 + t * 16 + m16) * K, 0, 3);

    v8f acc[2][4];
#pragma unroll
    for (int i = 0; i < 2; ++i)
#pragma unroll
        for (int t = 0; t < 4; ++t) acc[i][t] = (v8f){0, 0, 0, 0, 0, 0, 0, 0};

    for (int kb = 0; kb < K; kb += 32) {
        Frag16 af[2];
        load_frag(af[0], arow0 + kb, hi);
        load_frag(af[1], arow1 + kb, hi);
        Frag16 bf[4];
#pragma unroll
        for (int t = 0; t < 4; ++t)
            load_frag(bf[t], W + (size_t)(col0 + t * 16 + m16) * K + kb, hi);
#pragma unroll
        for (int i = 0; i < 2; ++i)
#pragma unroll
            for (int t = 0; t < 4; ++t)
                acc[i][t] = wmma_f16(af[i], bf[t], acc[i][t]);
    }

#pragma unroll
    for (int i = 0; i < 2; ++i)
#pragma unroll
        for (int t = 0; t < 4; ++t) {
            const int col = col0 + t * 16 + m16;
            const float badd = bias ? bias[col] : 0.f;
#pragma unroll
            for (int r = 0; r < 8; ++r) {
                const int row = row0 + i * 16 + r + hi * 8;
                C[(size_t)row * Cn + col] = acc[i][t][r] + badd;
            }
        }
}

// ---------------------------------------------------------------------------
// Flash attention: q[b][n][d] (f16, pre-scaled), k[b][m][d] (f16),
// vt[(b*H+h)*HD + c][m] (f16), mask[b][m] (int) -> out f32 [b][n][d].
// Block: 8 waves, each wave owns 16 query rows of one (b,h).
// Grid: (N/128, H, B).
// ---------------------------------------------------------------------------
__global__ __launch_bounds__(256)
void attn_wmma(const _Float16* __restrict__ q, const _Float16* __restrict__ k,
               const _Float16* __restrict__ vt, const int* __restrict__ mask,
               float* __restrict__ out) {
    const int lane = threadIdx.x & 31;
    const int wave = threadIdx.x >> 5;
    const int m16 = lane & 15, hi = lane >> 4;
    const int b = blockIdx.z, h = blockIdx.y;
    const int q0 = blockIdx.x * 128 + wave * 16;

    __shared__ _Float16 pshare[8][16 * 32];
    _Float16* pw = &pshare[wave][0];

    const _Float16* kbase = k + (size_t)b * M_ * D_ + h * HD;
    const _Float16* vbase = vt + ((size_t)(b * H_ + h) * HD) * M_;

    // One-shot warm-up prefetch of the first K/V tiles.
    __builtin_prefetch(kbase + (size_t)m16 * D_, 0, 3);
    __builtin_prefetch(kbase + (size_t)(16 + m16) * D_, 0, 3);
    __builtin_prefetch(vbase + (size_t)m16 * M_, 0, 3);
    __builtin_prefetch(vbase + (size_t)(32 + m16) * M_, 0, 3);

    // Q fragments: A[m][c], c = 0..63 -> two K=32 fragments.
    Frag16 af[2];
    {
        const _Float16* qrow = q + (size_t)(b * N_ + q0 + m16) * D_ + h * HD;
        load_frag(af[0], qrow, hi);
        load_frag(af[1], qrow + 32, hi);
    }

    v8f o[4];
#pragma unroll
    for (int t = 0; t < 4; ++t) o[t] = (v8f){0, 0, 0, 0, 0, 0, 0, 0};
    float mrow[8], lrow[8];
#pragma unroll
    for (int r = 0; r < 8; ++r) { mrow[r] = -3.0e38f; lrow[r] = 0.f; }

    for (int key0 = 0; key0 < M_; key0 += 32) {
        // ---- S = Q K^T for 32 keys (two 16-key tiles) ----
        v8f s[2];
#pragma unroll
        for (int t = 0; t < 2; ++t) {
            const _Float16* krow = kbase + (size_t)(key0 + 16 * t + m16) * D_;
            Frag16 bf0, bf1;
            load_frag(bf0, krow, hi);
            load_frag(bf1, krow + 32, hi);
            v8f a = (v8f){0, 0, 0, 0, 0, 0, 0, 0};
            a = wmma_f16(af[0], bf0, a);
            a = wmma_f16(af[1], bf1, a);
            s[t] = a;
        }
        // ---- mask (key index = lane's n) ----
#pragma unroll
        for (int t = 0; t < 2; ++t) {
            const int mv = mask[b * M_ + key0 + 16 * t + m16];
            if (mv == 0) {
#pragma unroll
                for (int r = 0; r < 8; ++r) s[t][r] = -1.0e30f;
            }
        }
        // ---- online softmax; rows m = r + 8*hi live in one 16-lane half ----
        float alpha[8];
#pragma unroll
        for (int r = 0; r < 8; ++r) {
            float tmax = fmaxf(s[0][r], s[1][r]);
#pragma unroll
            for (int off = 1; off < 16; off <<= 1)
                tmax = fmaxf(tmax, __shfl_xor(tmax, off, 32));
            const float mn = fmaxf(mrow[r], tmax);
            const float a = __expf(mrow[r] - mn);
            const float p0 = __expf(s[0][r] - mn);
            const float p1 = __expf(s[1][r] - mn);
            float rs = p0 + p1;
#pragma unroll
            for (int off = 1; off < 16; off <<= 1) rs += __shfl_xor(rs, off, 32);
            lrow[r] = lrow[r] * a + rs;
            mrow[r] = mn;
            alpha[r] = a;
            // stage P into per-wave LDS, row-major 16x32
            const int m = r + hi * 8;
            pw[m * 32 + m16]      = (_Float16)p0;
            pw[m * 32 + 16 + m16] = (_Float16)p1;
        }
#pragma unroll
        for (int t = 0; t < 4; ++t)
#pragma unroll
            for (int r = 0; r < 8; ++r) o[t][r] *= alpha[r];

        wave_lds_fence();
        // ---- P fragment (A-layout) from LDS ----
        Frag16 pf;
        load_frag(pf, pw + m16 * 32, hi);
        wave_lds_fence();

        // ---- O += P V : 4 column tiles of 16, K = 32 keys ----
#pragma unroll
        for (int t = 0; t < 4; ++t) {
            Frag16 vf;
            load_frag(vf, vbase + (size_t)(t * 16 + m16) * M_ + key0, hi);
            o[t] = wmma_f16(pf, vf, o[t]);
        }
    }

    // ---- epilogue: divide by l, store f32 [b][n][h*HD + c] ----
    float inv[8];
#pragma unroll
    for (int r = 0; r < 8; ++r) inv[r] = (lrow[r] > 0.f) ? (1.f / lrow[r]) : 0.f;
#pragma unroll
    for (int t = 0; t < 4; ++t)
#pragma unroll
        for (int r = 0; r < 8; ++r) {
            const int row = q0 + r + hi * 8;
            out[(size_t)(b * N_ + row) * D_ + h * HD + t * 16 + m16] =
                o[t][r] * inv[r];
        }
}

// ---------------------------------------------------------------------------
extern "C" void kernel_launch(void* const* d_in, const int* in_sizes, int n_in,
                              void* d_out, int out_size, void* d_ws, size_t ws_size,
                              hipStream_t stream) {
    (void)in_sizes; (void)n_in; (void)out_size; (void)ws_size;

    const float* querys    = (const float*)d_in[0];
    const float* key_feats = (const float*)d_in[1];
    const int*   mask      = (const int*)d_in[2];
    const float* Wq        = (const float*)d_in[3];
    const float* Wk        = (const float*)d_in[4];
    const float* Wv        = (const float*)d_in[5];
    const float* gq        = (const float*)d_in[6];
    const float* gk        = (const float*)d_in[7];
    const float* Wo        = (const float*)d_in[8];
    const float* bo        = (const float*)d_in[9];
    float*       out       = (float*)d_out;

    const size_t BND = (size_t)B_ * N_ * D_;  // 4,194,304
    const size_t DD  = (size_t)D_ * D_;       // 1,048,576
    const int    R   = B_ * N_;               // 4096 rows for all GEMMs

    char* ws = (char*)d_ws;
    size_t off = 0;
    auto alloc = [&](size_t bytes) -> void* {
        void* p = ws + off;
        off = (off + bytes + 255) & ~(size_t)255;
        return p;
    };

    _Float16* h_x   = (_Float16*)alloc(BND * 2);  // f16 activations (querys, later attn out)
    _Float16* h_kin = (_Float16*)alloc(BND * 2);  // f16 key_feats
    _Float16* h_Wq  = (_Float16*)alloc(DD * 2);
    _Float16* h_Wk  = (_Float16*)alloc(DD * 2);
    _Float16* h_Wv  = (_Float16*)alloc(DD * 2);
    _Float16* h_Wo  = (_Float16*)alloc(DD * 2);
    _Float16* h_q   = (_Float16*)alloc(BND * 2);  // rmsnorm'd + scaled q
    _Float16* h_k   = (_Float16*)alloc(BND * 2);  // rmsnorm'd k
    _Float16* h_vt  = (_Float16*)alloc(BND * 2);  // transposed v
    float*    f_prj = (float*)alloc(BND * 4);     // reused f32 staging

    const dim3 blk(256);
    const dim3 gcv(2048);
    const dim3 ggemm(R / 128, D_ / 128);          // (32, 8)
    const dim3 gattn(N_ / 128, H_, B_);           // (16, 16, 2)

    // 1) precision conversion
    cvt_f32_f16<<<gcv, blk, 0, stream>>>(querys,    h_x,   BND);
    cvt_f32_f16<<<gcv, blk, 0, stream>>>(key_feats, h_kin, BND);
    cvt_f32_f16<<<512, blk, 0, stream>>>(Wq, h_Wq, DD);
    cvt_f32_f16<<<512, blk, 0, stream>>>(Wk, h_Wk, DD);
    cvt_f32_f16<<<512, blk, 0, stream>>>(Wv, h_Wv, DD);
    cvt_f32_f16<<<512, blk, 0, stream>>>(Wo, h_Wo, DD);

    // 2) q = rmsnorm(querys @ Wq^T, gq) * HEAD_DIM^-0.5
    gemm_nt_wmma<<<ggemm, blk, 0, stream>>>(h_x, h_Wq, nullptr, f_prj, R, D_, D_);
    rmsnorm_cvt<<<R, blk, 0, stream>>>(f_prj, gq, 0.125f, h_q, D_);

    // 3) k = rmsnorm(key_feats @ Wk^T, gk)
    gemm_nt_wmma<<<ggemm, blk, 0, stream>>>(h_kin, h_Wk, nullptr, f_prj, R, D_, D_);
    rmsnorm_cvt<<<R, blk, 0, stream>>>(f_prj, gk, 1.0f, h_k, D_);

    // 4) v = key_feats @ Wv^T  (stored transposed for P@V fragment loads)
    gemm_nt_wmma<<<ggemm, blk, 0, stream>>>(h_kin, h_Wv, nullptr, f_prj, R, D_, D_);
    v_transpose_cvt<<<gcv, blk, 0, stream>>>(f_prj, h_vt);

    // 5) flash attention -> f32 x[b][n][d]
    attn_wmma<<<gattn, blk, 0, stream>>>(h_q, h_k, h_vt, mask, f_prj);

    // 6) out = x @ Wo^T + bo
    cvt_f32_f16<<<gcv, blk, 0, stream>>>(f_prj, h_x, BND);
    gemm_nt_wmma<<<ggemm, blk, 0, stream>>>(h_x, h_Wo, bo, out, R, D_, D_);
}
